// NTM_26190710571536
// MI455X (gfx1250) — compile-verified, matrix operand order checked by reading, not run on hardware
//
#include <hip/hip_runtime.h>
#include <hip/hip_bf16.h>

// ---------------- problem constants ----------------
#define BB 128   // batch
#define TT 512   // time steps
#define II 512   // input dim
#define HH 512   // hidden dim
#define OO 512   // output dim
#define NS 128   // memory slots
#define MD 20    // memory width
#define RHD 4    // read heads
#define BTILE 16 // batch rows per workgroup (== WMMA M)

// padded dims for WMMA tiling
#define WHN 80   // write-head outputs 66 -> 80  (5 N-tiles)
#define RHN 128  // read-head outputs 4*26 -> 4*32 (8 N-tiles)
#define RKP 96   // read vector 80 -> 96 (3 K-tiles)

// f16 weight scratch layout (element offsets in halves)
#define OFF_WIN   0
#define SZ_WIN    (HH*II)          // Wt_in  [o][i]
#define OFF_WREAD (OFF_WIN+SZ_WIN)
#define SZ_WREAD  (HH*RKP)         // Wt_read[o][k] k padded to 96
#define OFF_WOUT  (OFF_WREAD+SZ_WREAD)
#define SZ_WOUT   (OO*HH)          // Wt_out [o][h]
#define OFF_WWH   (OFF_WOUT+SZ_WOUT)
#define SZ_WWH    (WHN*HH)         // Wt_whead[n][h] n padded to 80
#define OFF_WRH   (OFF_WWH+SZ_WWH)
#define SZ_WRH    (RHN*HH)         // Wt_rhead[n][h] n padded to 128
#define TOTAL_HALVES (OFF_WRH+SZ_WRH)        // 679936
#define MEM_BYTE_OFF ((size_t)TOTAL_HALVES*2) // f32 memory state follows

typedef __attribute__((ext_vector_type(16))) _Float16 v16h;
typedef __attribute__((ext_vector_type(8)))  _Float16 v8h;
typedef __attribute__((ext_vector_type(4)))  _Float16 v4h;
typedef __attribute__((ext_vector_type(8)))  float    v8f;
typedef __attribute__((ext_vector_type(4)))  float    v4f;

// ---------------- helpers ----------------
__device__ __forceinline__ float sigmf_(float x)  { return 1.f / (1.f + expf(-x)); }
__device__ __forceinline__ float softpf_(float x) { return (x > 20.f) ? x : log1pf(expf(x)); }

// Load a 16-element f16 WMMA fragment (A row-major, or B from [N][K]-transposed
// weights): two contiguous 16-byte chunks per lane (CDNA5 16-bit A/B layout).
__device__ __forceinline__ v16h frag16(const _Float16* rowp, int kb, int lc) {
  const _Float16* p = rowp + kb + lc * 8;
  v8h lo = *(const v8h*)(p);
  v8h hi = *(const v8h*)(p + 16);
  v16h f;
#pragma unroll
  for (int i = 0; i < 8; ++i) { f[i] = lo[i]; f[i + 8] = hi[i]; }
  return f;
}

#define WMMA_F16(a, b, c) \
  __builtin_amdgcn_wmma_f32_16x16x32_f16(false, (a), false, (b), (short)0, (c), false, false)

// Async DMA of 16 bytes/lane from global to LDS (GV mode, ASYNCcnt-tracked).
__device__ __forceinline__ void async_g2l_b128(unsigned lds_off, const void* gaddr) {
  asm volatile("global_load_async_to_lds_b128 %0, %1, off"
               :: "v"(lds_off), "v"((unsigned long long)(uintptr_t)gaddr)
               : "memory");
}
__device__ __forceinline__ void wait_async0() {
  asm volatile("s_wait_asynccnt 0x0" ::: "memory");
}

// ---------------- kernel 0: convert + transpose weights to f16 ----------------
__global__ void ntm_convert_weights(const float* __restrict__ W_in,
                                    const float* __restrict__ W_read,
                                    const float* __restrict__ W_out,
                                    const float* __restrict__ W_rhead,
                                    const float* __restrict__ W_whead,
                                    _Float16* __restrict__ wt) {
  int idx = blockIdx.x * 256 + threadIdx.x;
  if (idx < SZ_WIN) {                       // Wt_in[o][i] = W_in[i][o]
    int o = idx >> 9, i = idx & 511;
    wt[OFF_WIN + idx] = (_Float16)W_in[i * HH + o];
  } else if (idx < OFF_WOUT) {              // Wt_read[o][k], K padded to 96
    int r = idx - OFF_WREAD;
    int n = r / RKP, k = r % RKP;
    wt[OFF_WREAD + r] = (_Float16)((k < RHD * MD) ? W_read[k * HH + n] : 0.f);
  } else if (idx < OFF_WWH) {               // Wt_out[o][h] = W_out[h][o]
    int r = idx - OFF_WOUT;
    int n = r >> 9, k = r & 511;
    wt[OFF_WOUT + r] = (_Float16)W_out[k * OO + n];
  } else if (idx < OFF_WRH) {               // Wt_whead[n][h], N padded to 80
    int r = idx - OFF_WWH;
    int n = r >> 9, k = r & 511;
    wt[OFF_WWH + r] = (_Float16)((n < 3 * MD + 6) ? W_whead[k * (3 * MD + 6) + n] : 0.f);
  } else if (idx < TOTAL_HALVES) {          // Wt_rhead[hd*32+o][h], o padded to 32
    int r = idx - OFF_WRH;
    int n = r >> 9, k = r & 511;
    int hd = n >> 5, o = n & 31;
    wt[OFF_WRH + r] = (_Float16)((o < MD + 6) ? W_rhead[(hd * HH + k) * (MD + 6) + o] : 0.f);
  }
}

// ---------------- addressing (one head) ----------------
__device__ void address_head(const float* hv, int hstride, int koff,
                             float* w_state, int wstride,
                             const float* memb,
                             float* wg, float* red, int tid) {
  const int b = tid >> 4, l16 = tid & 15;
  const float* hb = hv + b * hstride + koff;

  float kv[MD];
  float kn2 = 0.f;
#pragma unroll
  for (int m = 0; m < MD; ++m) { kv[m] = tanhf(hb[m]); kn2 += kv[m] * kv[m]; }
  float kn    = sqrtf(kn2);
  float beta  = softpf_(hb[20]);
  float g     = sigmf_(hb[21]);
  float s0r = hb[22], s1r = hb[23], s2r = hb[24];
  float smx = fmaxf(s0r, fmaxf(s1r, s2r));
  float e0 = expf(s0r - smx), e1 = expf(s1r - smx), e2 = expf(s2r - smx);
  float esi = 1.f / (e0 + e1 + e2);
  float s0 = e0 * esi, s1 = e1 * esi, s2 = e2 * esi;
  float gamma = 1.f + softpf_(hb[25]);

  float av[8];
  float lmax = -3.4e38f;
#pragma unroll
  for (int jj = 0; jj < 8; ++jj) {
    int n = l16 * 8 + jj;
    const float* mrow = memb + (b * NS + n) * MD;
    float dot = 0.f, mn2 = 0.f;
#pragma unroll
    for (int m = 0; m < MD; ++m) { float mv = mrow[m]; dot += mv * kv[m]; mn2 += mv * mv; }
    float sim = dot / (sqrtf(mn2) * kn + 1e-8f);
    av[jj] = beta * sim;
    lmax = fmaxf(lmax, av[jj]);
  }
  red[b * 16 + l16] = lmax;
  __syncthreads();
  float mx = -3.4e38f;
  for (int i = 0; i < 16; ++i) mx = fmaxf(mx, red[b * 16 + i]);
  float lsum = 0.f;
#pragma unroll
  for (int jj = 0; jj < 8; ++jj) { av[jj] = expf(av[jj] - mx); lsum += av[jj]; }
  __syncthreads();
  red[b * 16 + l16] = lsum;
  __syncthreads();
  float tot = 0.f;
  for (int i = 0; i < 16; ++i) tot += red[b * 16 + i];
  float inv = 1.f / tot;
#pragma unroll
  for (int jj = 0; jj < 8; ++jj) {
    int n = l16 * 8 + jj;
    float wc = av[jj] * inv;
    wg[b * NS + n] = g * wc + (1.f - g) * w_state[b * wstride + n];
  }
  __syncthreads();
  float wp[8];
  float lps = 0.f;
#pragma unroll
  for (int jj = 0; jj < 8; ++jj) {
    int n = l16 * 8 + jj;
    float wsv = s0 * wg[b * NS + ((n + 1) & (NS - 1))] +
                s1 * wg[b * NS + n] +
                s2 * wg[b * NS + ((n - 1) & (NS - 1))];
    wp[jj] = powf(wsv, gamma);
    lps += wp[jj];
  }
  __syncthreads();
  red[b * 16 + l16] = lps;
  __syncthreads();
  float ps = 0.f;
  for (int i = 0; i < 16; ++i) ps += red[b * 16 + i];
  float pinv = 1.f / (ps + 1e-8f);
#pragma unroll
  for (int jj = 0; jj < 8; ++jj) {
    int n = l16 * 8 + jj;
    w_state[b * wstride + n] = wp[jj] * pinv;
  }
  __syncthreads();
}

// ---------------- kernel 1: full recurrence, 16 batch rows per WG ----------------
__global__ __launch_bounds__(256, 1) void ntm_recurrent(
    const float* __restrict__ x,
    const float* __restrict__ b_in, const float* __restrict__ b_read,
    const float* __restrict__ b_out,
    const float* __restrict__ b_rhead, const float* __restrict__ b_whead,
    const _Float16* __restrict__ wt,
    float* __restrict__ memw,      // [BB][NS][MD] f32 state in scratch (L2 resident)
    float* __restrict__ out)       // [BB][TT][OO]
{
  const _Float16* Wt_in   = wt + OFF_WIN;
  const _Float16* Wt_read = wt + OFF_WREAD;
  const _Float16* Wt_out  = wt + OFF_WOUT;
  const _Float16* Wt_wh   = wt + OFF_WWH;
  const _Float16* Wt_rh   = wt + OFF_WRH;

  const int tid  = threadIdx.x;
  const int wv   = tid >> 5;           // wave id 0..7
  const int lane = tid & 31;
  const int l15  = lane & 15;
  const int lc   = lane >> 4;          // 0/1 half-wave selector
  const int b0   = blockIdx.x * BTILE; // first batch row for this WG

  // LDS state (~162 KB, within CDNA5's 320 KB/WGP)
  __shared__ float    x32s[2][BTILE * II];    // async-DMA double buffer (raw f32)
  __shared__ _Float16 h16[BTILE * 520];       // h,  f16, stride 520 (bank-skewed)
  __shared__ _Float16 x16[BTILE * 520];       // x_t f16
  __shared__ _Float16 r16[BTILE * RKP];       // read vectors, K-padded
  __shared__ float    wh_ls[BTILE * WHN];     // write-head params
  __shared__ float    rh_ls[BTILE * RHN];     // read-head params
  __shared__ float    ww_ls[BTILE * NS];      // write weights state
  __shared__ float    wr_ls[BTILE * RHD * NS];// read weights state
  __shared__ float    wg_ls[BTILE * NS];      // addressing scratch
  __shared__ float    red[256];               // reduction scratch

  // ---- init state ----
  for (int i = tid; i < BTILE * 520; i += 256) { h16[i] = (_Float16)0.f; x16[i] = (_Float16)0.f; }
  for (int i = tid; i < BTILE * NS; i += 256) ww_ls[i] = 1.f / NS;
  for (int i = tid; i < BTILE * RHD * NS; i += 256) wr_ls[i] = 1.f / NS;
  for (int i = tid; i < BTILE * NS * MD; i += 256) {
    memw[(size_t)b0 * NS * MD + i] = 1e-6f;
  }

  // prime the async x pipeline: DMA x_0 into buffer 0 (self-produced chunks)
  for (int i = tid; i < BTILE * II / 4; i += 256) {
    int b = i >> 7, col = (i & 127) * 4;
    async_g2l_b128((unsigned)(uintptr_t)&x32s[0][b * II + col],
                   x + ((size_t)(b0 + b) * TT + 0) * II + col);
  }
  __syncthreads();

  const float* memb = memw + (size_t)b0 * NS * MD;
  int pb = 0;

  for (int t = 0; t < TT; ++t) {
    // ---- phase 0: convert staged x_t (f32 LDS -> f16 LDS); DMA x_{t+1} ----
    wait_async0();  // this wave's DMA chunks (issued last step) have landed
    for (int i = tid; i < BTILE * II / 4; i += 256) {
      int b = i >> 7, col = (i & 127) * 4;
      v4f xv = *(const v4f*)&x32s[pb][b * II + col];
      v4h hv;
#pragma unroll
      for (int q = 0; q < 4; ++q) hv[q] = (_Float16)xv[q];
      *(v4h*)&x16[b * 520 + col] = hv;
    }
    if (t + 1 < TT) {
      for (int i = tid; i < BTILE * II / 4; i += 256) {
        int b = i >> 7, col = (i & 127) * 4;
        async_g2l_b128((unsigned)(uintptr_t)&x32s[pb ^ 1][b * II + col],
                       x + ((size_t)(b0 + b) * TT + (t + 1)) * II + col);
      }
    }
    __syncthreads();

    // ---- phase 1: wh = h@W_whead, rh = h@W_rhead (13 N-tiles over 8 waves),
    //      software-pipelined K loop
    for (int tile = wv; tile < 13; tile += 8) {
      v8f c = {};
      int col0 = (tile < 5) ? tile * 16 : (tile - 5) * 16;
      const _Float16* brow = ((tile < 5) ? Wt_wh : Wt_rh) + (col0 + l15) * HH;
      const _Float16* arow = h16 + l15 * 520;
      v16h a = frag16(arow, 0, lc);
      v16h bf = frag16(brow, 0, lc);
#pragma unroll 1
      for (int kb = 0; kb < HH; kb += 32) {
        int kn = (kb + 32 < HH) ? kb + 32 : 0;
        v16h an = frag16(arow, kn, lc);
        v16h bn = frag16(brow, kn, lc);
        c = WMMA_F16(a, bf, c);
        a = an; bf = bn;
      }
      int col = col0 + l15;
      if (tile < 5) {
        float bias = (col < 3 * MD + 6) ? b_whead[col] : 0.f;
#pragma unroll
        for (int j = 0; j < 8; ++j) wh_ls[(lc * 8 + j) * WHN + col] = c[j] + bias;
      } else {
        int hd = col >> 5, o = col & 31;
        float bias = (o < MD + 6) ? b_rhead[hd * (MD + 6) + o] : 0.f;
#pragma unroll
        for (int j = 0; j < 8; ++j) rh_ls[(lc * 8 + j) * RHN + col] = c[j] + bias;
      }
    }

    // ---- phase 1b: xacc = x_t @ W_in (registers, pipelined, A reused 4x) ----
    v8f xacc[4] = {};
    {
      const _Float16* arow = x16 + l15 * 520;
      const _Float16* brow = Wt_in + (wv * 64 + l15) * II;
      v16h a = frag16(arow, 0, lc);
      v16h bf[4];
#pragma unroll
      for (int j = 0; j < 4; ++j) bf[j] = frag16(brow + j * 16 * II, 0, lc);
#pragma unroll 1
      for (int kb = 0; kb < II; kb += 32) {
        int kn = (kb + 32 < II) ? kb + 32 : 0;
        v16h an = frag16(arow, kn, lc);
        v16h bn[4];
#pragma unroll
        for (int j = 0; j < 4; ++j) bn[j] = frag16(brow + j * 16 * II, kn, lc);
#pragma unroll
        for (int j = 0; j < 4; ++j) xacc[j] = WMMA_F16(a, bf[j], xacc[j]);
        a = an;
#pragma unroll
        for (int j = 0; j < 4; ++j) bf[j] = bn[j];
      }
    }
    __syncthreads();

    // ---- phase 2: write-head addressing + memory update ----
    address_head(wh_ls, WHN, 0, ww_ls, NS, memb, wg_ls, red, tid);
    for (int i = tid; i < BTILE * NS * MD; i += 256) {
      int b = i / (NS * MD), rem = i % (NS * MD), n = rem / MD, m = rem % MD;
      float wvv = ww_ls[b * NS + n];
      float er  = sigmf_(wh_ls[b * WHN + MD + 6 + m]);
      float ad  = tanhf(wh_ls[b * WHN + 2 * MD + 6 + m]);
      float* p  = memw + (((size_t)(b0 + b) * NS + n) * MD + m);
      *p = *p * (1.f - wvv * er) + wvv * ad;
    }
    __syncthreads();

    // ---- phase 3: read-head addressing (updated memory), then read vectors ----
    for (int hd = 0; hd < RHD; ++hd)
      address_head(rh_ls, RHN, hd * 32, wr_ls + hd * NS, RHD * NS, memb, wg_ls, red, tid);

    for (int i = tid; i < BTILE * RKP; i += 256) {
      int b = i / RKP, k = i % RKP;
      _Float16 val = (_Float16)0.f;
      if (k < RHD * MD) {
        int hd = k / MD, m = k % MD;
        float acc = 0.f;
        for (int n = 0; n < NS; ++n)
          acc += wr_ls[(b * RHD + hd) * NS + n] * memb[(b * NS + n) * MD + m];
        val = (_Float16)acc;
      }
      r16[b * RKP + k] = val;
    }
    __syncthreads();

    // ---- phase 4: h = relu(xacc + r @ W_read + b_in + b_read), pipelined ----
    {
      const _Float16* arow = r16 + l15 * RKP;
      const _Float16* brow = Wt_read + (wv * 64 + l15) * RKP;
      v16h a = frag16(arow, 0, lc);
      v16h bf[4];
#pragma unroll
      for (int j = 0; j < 4; ++j) bf[j] = frag16(brow + j * 16 * RKP, 0, lc);
#pragma unroll 1
      for (int kb = 0; kb < RKP; kb += 32) {
        int kn = (kb + 32 < RKP) ? kb + 32 : 0;
        v16h an = frag16(arow, kn, lc);
        v16h bn[4];
#pragma unroll
        for (int j = 0; j < 4; ++j) bn[j] = frag16(brow + j * 16 * RKP, kn, lc);
#pragma unroll
        for (int j = 0; j < 4; ++j) xacc[j] = WMMA_F16(a, bf[j], xacc[j]);
        a = an;
#pragma unroll
        for (int j = 0; j < 4; ++j) bf[j] = bn[j];
      }
#pragma unroll
      for (int j = 0; j < 4; ++j) {
        int col = (wv * 4 + j) * 16 + l15;
        float bias = b_in[col] + b_read[col];
#pragma unroll
        for (int j8 = 0; j8 < 8; ++j8)
          h16[(lc * 8 + j8) * 520 + col] = (_Float16)fmaxf(xacc[j][j8] + bias, 0.f);
      }
    }
    __syncthreads();

    // ---- phase 5: out = sigmoid(h @ W_out + b_out), pipelined ----
    {
      v8f c[4] = {};
      const _Float16* arow = h16 + l15 * 520;
      const _Float16* brow = Wt_out + (wv * 64 + l15) * HH;
      v16h a = frag16(arow, 0, lc);
      v16h bf[4];
#pragma unroll
      for (int j = 0; j < 4; ++j) bf[j] = frag16(brow + j * 16 * HH, 0, lc);
#pragma unroll 1
      for (int kb = 0; kb < HH; kb += 32) {
        int kn = (kb + 32 < HH) ? kb + 32 : 0;
        v16h an = frag16(arow, kn, lc);
        v16h bn[4];
#pragma unroll
        for (int j = 0; j < 4; ++j) bn[j] = frag16(brow + j * 16 * HH, kn, lc);
#pragma unroll
        for (int j = 0; j < 4; ++j) c[j] = WMMA_F16(a, bf[j], c[j]);
        a = an;
#pragma unroll
        for (int j = 0; j < 4; ++j) bf[j] = bn[j];
      }
#pragma unroll
      for (int j = 0; j < 4; ++j) {
        int col = (wv * 4 + j) * 16 + l15;
        float bo = b_out[col];
#pragma unroll
        for (int j8 = 0; j8 < 8; ++j8) {
          int row = lc * 8 + j8;
          out[((size_t)(b0 + row) * TT + t) * OO + col] = sigmf_(c[j][j8] + bo);
        }
      }
    }
    __syncthreads();
    pb ^= 1;
  }
}

// ---------------- host entry ----------------
extern "C" void kernel_launch(void* const* d_in, const int* in_sizes, int n_in,
                              void* d_out, int out_size, void* d_ws, size_t ws_size,
                              hipStream_t stream) {
  const float* x        = (const float*)d_in[0];
  const float* W_in     = (const float*)d_in[1];
  const float* b_in     = (const float*)d_in[2];
  const float* W_read   = (const float*)d_in[3];
  const float* b_read   = (const float*)d_in[4];
  const float* W_out    = (const float*)d_in[5];
  const float* b_out    = (const float*)d_in[6];
  const float* W_rhead  = (const float*)d_in[7];
  const float* b_rhead  = (const float*)d_in[8];
  const float* W_whead  = (const float*)d_in[9];
  const float* b_whead  = (const float*)d_in[10];

  _Float16* wt  = (_Float16*)d_ws;
  float* memw   = (float*)((char*)d_ws + MEM_BYTE_OFF);
  float* out    = (float*)d_out;

  int cvt_blocks = (TOTAL_HALVES + 255) / 256;
  ntm_convert_weights<<<cvt_blocks, 256, 0, stream>>>(W_in, W_read, W_out, W_rhead, W_whead, wt);

  ntm_recurrent<<<BB / BTILE, 256, 0, stream>>>(
      x, b_in, b_read, b_out, b_rhead, b_whead, wt, memw, out);
}
